// MANN_11948599017886
// MI455X (gfx1250) — matrix operand order repaired and checked by reading, not compiled, not run on hardware
//
#include <hip/hip_runtime.h>

// ---------------------------------------------------------------------------
// MANN mixture-of-experts forward, CDNA5 (gfx1250, wave32, WMMA + TDM).
// B=1024, X_MAIN=480, X_GATE=128, Y=640, HID=1024, GHID=64, K=8 experts.
//
// out[b,n] = sum_e sum_k (ew[b,e]*A[b,k]) * W[e,n,k]: the expert mix folds
// into the GEMM K loop by scaling the A fragment per expert (per-row scale ==
// per-lane uniform in the WMMA A layout).  All 8 experts' scaled A fragments
// are materialized in distinct registers per K-step (no WAR hazard NOPs
// around the WMMAs), and the 8-expert B slab is streamed by the Tensor Data
// Mover into a double-buffered LDS slab, software-pipelined one K-step ahead.
// ---------------------------------------------------------------------------

typedef __attribute__((ext_vector_type(16))) _Float16 v16h;
typedef __attribute__((ext_vector_type(8)))  _Float16 v8h;
typedef __attribute__((ext_vector_type(4)))  _Float16 v4h;
typedef __attribute__((ext_vector_type(8)))  float    v8f;
typedef __attribute__((ext_vector_type(4)))  unsigned int v4u;
typedef __attribute__((ext_vector_type(8)))  int      v8i;
typedef __attribute__((ext_vector_type(4)))  int      v4i;

#define B_DIM   1024
#define X_MAIN  480
#define X_GATE  128
#define Y_DIM   640
#define HID     1024
#define GHID    64
#define KEXP    8

#define TM 64          // workgroup output tile M
#define TN 64          // workgroup output tile N
#define TK 32          // K step per WMMA
#define PITCH 40       // LDS row pitch (f16 elems); 80B stride -> conflict-free b128
#define A_ELEMS (TM * PITCH)            // one A slab (f16 elems)
#define B_ELEMS (KEXP * TN * PITCH)     // one 8-expert B slab (f16 elems)

#if __has_builtin(__builtin_amdgcn_tensor_load_to_lds) && \
    __has_builtin(__builtin_amdgcn_s_wait_tensorcnt)
#define USE_TDM 1
#else
#define USE_TDM 0
#endif

__device__ __forceinline__ float eluf(float x) {
  return x > 0.0f ? x : __expf(x) - 1.0f;
}

__device__ __forceinline__ v8f wmma16(v16h a, v16h b, v8f c) {
  // (neg_a, A, neg_b, B, c_mod, C, reuse_a, reuse_b)
  return __builtin_amdgcn_wmma_f32_16x16x32_f16(false, a, false, b, (short)0, c,
                                                false, false);
}

// A-matrix 16x32 f16 fragment (ISA 7.12.2): lanes 0-15 row M=lane,
// regs 0-3 = K[0..7], regs 4-7 = K[16..23]; lanes 16-31 K[8..15]/K[24..31].
__device__ __forceinline__ v16h frag_a(const _Float16* ls, int row, int lhi) {
  union { v16h v; v8h h[2]; } u;
  const _Float16* p = ls + row * PITCH;
  u.h[0] = *(const v8h*)(p + lhi * 8);
  u.h[1] = *(const v8h*)(p + 16 + lhi * 8);
  return u.v;
}

// B-matrix 32x16 f16 fragment: lane n holds column n; lanes 0-15 K[0..7]/K[16..23],
// lanes 16-31 K[8..15]/K[24..31] (same K order as the A fragment).
__device__ __forceinline__ v16h frag_b(const _Float16* ls, int row, int lhi) {
  union { v16h v; v8h h[2]; } u;
  const _Float16* p = ls + row * PITCH + lhi * 8;
  u.h[0] = *(const v8h*)(p);
  u.h[1] = *(const v8h*)(p + 16);
  return u.v;
}

#if USE_TDM
// One TDM descriptor: load tile x=32 (contiguous K elems), y=64 (rows, stride
// Kd), z=8 (experts, stride N*Kd) of f16 data at gp into LDS at lds_off with
// a 4-DWORD pad after every 16-DWORD row (=> 40-elem LDS pitch).
__device__ __forceinline__ void tdm_load_b_slab(const _Float16* gp,
                                                unsigned lds_off, int Kd, int N) {
  unsigned long long ga = (unsigned long long)(size_t)gp;
  unsigned long long s0 = (unsigned long long)(unsigned)Kd;               // row stride
  unsigned long long s1 = (unsigned long long)(unsigned)N * (unsigned)Kd; // expert stride
  v4u g0;
  g0[0] = 1u;                                  // count=1 valid user descriptor
  g0[1] = lds_off;                             // LDS byte address
  g0[2] = (unsigned)ga;                        // global_addr[31:0]
  g0[3] = ((unsigned)(ga >> 32) & 0x01FFFFFFu) // global_addr[56:32]
          | (2u << 30);                        // type = 2 ("image")
  unsigned d0 = (unsigned)Kd;                  // tensor_dim0 (tiles in-bounds)
  unsigned d1 = 1u << 24;                      // tensor_dim1: large, in-bounds
  v8i g1;
  g1[0] = (int)((1u << 16)      // data_size = 1 -> 2 bytes
                | (1u << 20)    // pad_enable
                | (3u << 22)    // pad_interval: 16 DWORDs (one 32-elem row)
                | (3u << 25));  // pad_amount: 4 DWORDs (8 f16 elems)
  g1[1] = (int)((d0 & 0xFFFFu) << 16);                      // tensor_dim0[15:0]
  g1[2] = (int)((d0 >> 16) | ((d1 & 0xFFFFu) << 16));       // dim0[31:16]|dim1[15:0]
  g1[3] = (int)((d1 >> 16) | (32u << 16));                  // dim1[31:16]|tile_dim0=32
  g1[4] = (int)(64u | (8u << 16));                          // tile_dim1=64, tile_dim2=8
  g1[5] = (int)(unsigned)s0;                                // dim0_stride[31:0]
  g1[6] = (int)(((unsigned)(s0 >> 32) & 0xFFFFu)            // dim0_stride[47:32]
                | (((unsigned)s1 & 0xFFFFu) << 16));        // dim1_stride[15:0]
  g1[7] = (int)(unsigned)(s1 >> 16);                        // dim1_stride[47:16]
  v4i g2 = { 8, 0, 0, 0 };   // tensor_dim2 = 8 experts; tile_dim3 = 0 (2-3D only)
  v4i g3 = { 0, 0, 0, 0 };
  v8i g4 = { 0, 0, 0, 0, 0, 0, 0, 0 };  // extra group in 6-arg form: unused
  __builtin_amdgcn_tensor_load_to_lds(g0, g1, g2, g3, g4, 0);
}
#endif

// ---------------------------------------------------------------------------
// Fused expert layer:
//   out[b,n] = act( sum_e ew[b,e]*(A[b,:].W[e,n,:]) + sum_e ew[b,e]*bias[e,n] )
// A: f16 [M,Kd] row-major, W: f16 [8,N,Kd] row-major (GEMM vs W^T).
// 128 threads = 4 waves; each wave owns a 32x32 subtile (2x2 WMMA frags) and
// accumulates all 8 experts into one accumulator set via per-lane A scaling.
// LDS double-buffered; TDM prefetches the next K-step's B slab during compute.
// ---------------------------------------------------------------------------
__global__ __launch_bounds__(128)
void expert_layer_wmma(const _Float16* __restrict__ A,
                       const _Float16* __restrict__ W,
                       const float*    __restrict__ ew,    // [M,8]
                       const float*    __restrict__ bias,  // [8,N]
                       _Float16*       __restrict__ out_h, // f16 out (or null)
                       float*          __restrict__ out_f, // f32 out (or null)
                       int M, int N, int Kd, int do_elu) {
  __shared__ _Float16 Als[2 * A_ELEMS];   // 2 x 5 KB
  __shared__ _Float16 Bls[2 * B_ELEMS];   // 2 x 40 KB: double-buffered 8-expert slab
  __shared__ float    ewls[TM * KEXP];    // 2 KB
  __shared__ float    bls[KEXP * TN];     // 2 KB

  const int tid   = threadIdx.x;
  const int lane  = tid & 31;
  const int wave  = tid >> 5;     // 0..3
  const int wm    = wave >> 1;    // 0..1  (M half)
  const int wn    = wave & 1;     // 0..1  (N half)
  const int lrow  = lane & 15;
  const int lhi   = lane >> 4;    // 0/1
  const int m_blk = blockIdx.y * TM;
  const int n_blk = blockIdx.x * TN;
  const int wuni  = __builtin_amdgcn_readfirstlane(wave);  // provably uniform
  const _Float16* Wblk = W + (size_t)n_blk * Kd;

#if USE_TDM
  // Kick off the first B slab DMA as early as possible.
  if (wuni == 0)
    tdm_load_b_slab(Wblk, (unsigned)(size_t)(void*)&Bls[0], Kd, N);
#endif

  // Stage expert weights and biases for this tile.
  for (int i = tid; i < TM * KEXP; i += 128)
    ewls[i] = ew[(size_t)(m_blk + (i >> 3)) * KEXP + (i & 7)];
  for (int i = tid; i < KEXP * TN; i += 128)
    bls[i] = bias[(size_t)(i >> 6) * N + n_blk + (i & 63)];
  __syncthreads();

  // Per-lane expert scales for the two A fragment rows this lane holds.
  _Float16 s0h[KEXP], s1h[KEXP];
#pragma unroll
  for (int e = 0; e < KEXP; ++e) {
    s0h[e] = (_Float16)ewls[(wm * 32 + lrow) * KEXP + e];
    s1h[e] = (_Float16)ewls[(wm * 32 + 16 + lrow) * KEXP + e];
  }

  v8f acc00 = {}, acc01 = {}, acc10 = {}, acc11 = {};

  int ib = 0;
  for (int kk = 0; kk < Kd; kk += TK, ib ^= 1) {
    _Float16* Ab = Als + ib * A_ELEMS;
    _Float16* Bb = Bls + ib * B_ELEMS;

    // A tile 64x32 into slab ib: 256 8-elem chunks, 2 per thread.
    // Safe: the last reads of slab ib finished before the previous barrier.
#pragma unroll
    for (int i = 0; i < 2; ++i) {
      int c = tid + i * 128;
      int r = c >> 2, col = (c & 3) * 8;
      *(v8h*)&Ab[r * PITCH + col] =
          *(const v8h*)&A[(size_t)(m_blk + r) * Kd + kk + col];
    }
    if (kk + TK < Kd) {  // global_prefetch_b8 next A slab
      int r = tid >> 1, col = (tid & 1) * 16;
      __builtin_prefetch(&A[(size_t)(m_blk + r) * Kd + kk + TK + col], 0, 1);
    }

#if !USE_TDM
    // Fallback: cooperative vector copy of the 8-expert slab into slab ib.
#pragma unroll
    for (int i = 0; i < 16; ++i) {
      int c = tid + i * 128;
      int e = c >> 8, r = (c >> 2) & 63, col = (c & 3) * 8;
      *(v8h*)&Bb[(e * TN + r) * PITCH + col] =
          *(const v8h*)&W[((size_t)e * N + n_blk + r) * Kd + kk + col];
    }
#else
    if (wuni == 0) __builtin_amdgcn_s_wait_tensorcnt((short)0);  // slab ib ready
#endif
    __syncthreads();  // publish TDM slab + A stores; all lanes done reading ib^1

#if USE_TDM
    // Pipeline: start the DMA for the next K-step into the other slab while
    // this step computes.  Safe: every lane passed the barrier above, so all
    // reads of slab ib^1 (previous iteration's compute) have completed.
    if (wuni == 0 && kk + TK < Kd)
      tdm_load_b_slab(Wblk + (kk + TK),
                      (unsigned)(size_t)(void*)&Bls[(ib ^ 1) * B_ELEMS], Kd, N);
#endif

    // Pre-scale both A fragments for all 8 experts into distinct registers:
    // no VALU writes near the WMMA operand regs -> no WAR hazard NOPs.
    v16h a0 = frag_a(Ab, wm * 32 + lrow, lhi);
    v16h a1 = frag_a(Ab, wm * 32 + 16 + lrow, lhi);
    v16h as0[KEXP], as1[KEXP];
#pragma unroll
    for (int e = 0; e < KEXP; ++e) {
      as0[e] = a0 * s0h[e];   // v_pk_mul_f16 x8
      as1[e] = a1 * s1h[e];
    }
#pragma unroll
    for (int e = 0; e < KEXP; ++e) {
      const _Float16* bl = Bb + e * (TN * PITCH);
      v16h b0 = frag_b(bl, wn * 32 + lrow, lhi);
      v16h b1 = frag_b(bl, wn * 32 + 16 + lrow, lhi);
      acc00 = wmma16(as0[e], b0, acc00);
      acc01 = wmma16(as0[e], b1, acc01);
      acc10 = wmma16(as1[e], b0, acc10);
      acc11 = wmma16(as1[e], b1, acc11);
    }
  }

  // Epilogue: mixed bias + activation + store.
  const int nl0 = wn * 32 + lrow;
  const int nl1 = nl0 + 16;
#pragma unroll
  for (int v = 0; v < 8; ++v) {
    int r0l = wm * 32 + v + 8 * lhi;   // C/D layout: M = v + 8*(lane>>4)
    int r1l = r0l + 16;
    float bm00 = 0.f, bm01 = 0.f, bm10 = 0.f, bm11 = 0.f;
#pragma unroll
    for (int e2 = 0; e2 < KEXP; ++e2) {
      float we0 = ewls[r0l * KEXP + e2];
      float we1 = ewls[r1l * KEXP + e2];
      float bc0 = bls[e2 * TN + nl0];
      float bc1 = bls[e2 * TN + nl1];
      bm00 += we0 * bc0; bm01 += we0 * bc1;
      bm10 += we1 * bc0; bm11 += we1 * bc1;
    }
    float v00 = acc00[v] + bm00, v01 = acc01[v] + bm01;
    float v10 = acc10[v] + bm10, v11 = acc11[v] + bm11;
    if (do_elu) { v00 = eluf(v00); v01 = eluf(v01); v10 = eluf(v10); v11 = eluf(v11); }

    size_t m0 = (size_t)(m_blk + r0l), m1 = (size_t)(m_blk + r1l);
    int ng0 = n_blk + nl0, ng1 = n_blk + nl1;
    if (out_h) {
      out_h[m0 * N + ng0] = (_Float16)v00;
      out_h[m0 * N + ng1] = (_Float16)v01;
      out_h[m1 * N + ng0] = (_Float16)v10;
      out_h[m1 * N + ng1] = (_Float16)v11;
    } else {
      out_f[m0 * N + ng0] = v00;
      out_f[m0 * N + ng1] = v01;
      out_f[m1 * N + ng0] = v10;
      out_f[m1 * N + ng1] = v11;
    }
  }
}

// ---------------------------------------------------------------------------
// Small gating network (negligible FLOPs -> plain VALU kernels).
// ---------------------------------------------------------------------------
__global__ void gate_fc_elu(const float* __restrict__ x, const float* __restrict__ w,
                            const float* __restrict__ b, float* __restrict__ out,
                            int In, int Out) {
  __shared__ float xs[128];
  const int bi = blockIdx.x;
  for (int i = threadIdx.x; i < In; i += blockDim.x)
    xs[i] = x[(size_t)bi * In + i];
  __syncthreads();
  int j = threadIdx.x;
  if (j < Out) {
    float s = b[j];
    const float* wr = w + (size_t)j * In;
    for (int i = 0; i < In; ++i) s += xs[i] * wr[i];
    out[(size_t)bi * Out + j] = eluf(s);
  }
}

__global__ void gate_softmax(const float* __restrict__ g, const float* __restrict__ w,
                             const float* __restrict__ b, float* __restrict__ ew) {
  int bi = blockIdx.x * blockDim.x + threadIdx.x;
  if (bi >= B_DIM) return;
  const float* gr = g + (size_t)bi * GHID;
  float logit[KEXP];
  float mx = -1e30f;
#pragma unroll
  for (int k = 0; k < KEXP; ++k) {
    float s = b[k];
    const float* wr = w + (size_t)k * GHID;
    for (int i = 0; i < GHID; ++i) s += gr[i] * wr[i];
    logit[k] = s;
    mx = fmaxf(mx, s);
  }
  float den = 0.f;
#pragma unroll
  for (int k = 0; k < KEXP; ++k) { logit[k] = __expf(logit[k] - mx); den += logit[k]; }
  float inv = 1.f / den;
#pragma unroll
  for (int k = 0; k < KEXP; ++k) ew[(size_t)bi * KEXP + k] = logit[k] * inv;
}

// f32 -> f16 conversion (element counts are multiples of 4).
__global__ void cvt_f32_f16(const float* __restrict__ in, _Float16* __restrict__ out,
                            int n4) {
  int i = blockIdx.x * blockDim.x + threadIdx.x;
  if (i < n4) {
    float4 f = ((const float4*)in)[i];
    v4h h = { (_Float16)f.x, (_Float16)f.y, (_Float16)f.z, (_Float16)f.w };
    ((v4h*)out)[i] = h;
  }
}

// ---------------------------------------------------------------------------
extern "C" void kernel_launch(void* const* d_in, const int* in_sizes, int n_in,
                              void* d_out, int out_size, void* d_ws, size_t ws_size,
                              hipStream_t stream) {
  (void)in_sizes; (void)n_in; (void)out_size; (void)ws_size;
  const float* x_main = (const float*)d_in[0];
  const float* x_gate = (const float*)d_in[1];
  const float* g1_w   = (const float*)d_in[2];
  const float* g1_b   = (const float*)d_in[3];
  const float* g2_w   = (const float*)d_in[4];
  const float* g2_b   = (const float*)d_in[5];
  const float* g3_w   = (const float*)d_in[6];
  const float* g3_b   = (const float*)d_in[7];
  const float* W1     = (const float*)d_in[8];
  const float* b1     = (const float*)d_in[9];
  const float* W2     = (const float*)d_in[10];
  const float* b2     = (const float*)d_in[11];
  const float* W3     = (const float*)d_in[12];
  const float* b3     = (const float*)d_in[13];
  float* y = (float*)d_out;

  // Carve scratch (256B-aligned slabs).
  char* wsp = (char*)d_ws;
  auto carve = [&](size_t bytes) {
    char* p = wsp;
    wsp += (bytes + 255) & ~(size_t)255;
    return p;
  };
  _Float16* xh  = (_Float16*)carve((size_t)B_DIM * X_MAIN * 2);
  _Float16* W1h = (_Float16*)carve((size_t)KEXP * HID * X_MAIN * 2);
  _Float16* W2h = (_Float16*)carve((size_t)KEXP * HID * HID * 2);
  _Float16* W3h = (_Float16*)carve((size_t)KEXP * Y_DIM * HID * 2);
  float*    g1o = (float*)carve((size_t)B_DIM * GHID * 4);
  float*    g2o = (float*)carve((size_t)B_DIM * GHID * 4);
  float*    ewb = (float*)carve((size_t)B_DIM * KEXP * 4);
  _Float16* h1  = (_Float16*)carve((size_t)B_DIM * HID * 2);
  _Float16* h2  = (_Float16*)carve((size_t)B_DIM * HID * 2);

  // 1) downconvert activations + expert weights to f16
  auto cvt = [&](const float* src, _Float16* dst, size_t n) {
    int n4 = (int)(n / 4);
    cvt_f32_f16<<<(n4 + 255) / 256, 256, 0, stream>>>(src, dst, n4);
  };
  cvt(x_main, xh,  (size_t)B_DIM * X_MAIN);
  cvt(W1,     W1h, (size_t)KEXP * HID * X_MAIN);
  cvt(W2,     W2h, (size_t)KEXP * HID * HID);
  cvt(W3,     W3h, (size_t)KEXP * Y_DIM * HID);

  // 2) gating network -> expert weights [B,8]
  gate_fc_elu<<<B_DIM, GHID, 0, stream>>>(x_gate, g1_w, g1_b, g1o, X_GATE, GHID);
  gate_fc_elu<<<B_DIM, GHID, 0, stream>>>(g1o,   g2_w, g2_b, g2o, GHID,  GHID);
  gate_softmax<<<B_DIM / 256, 256, 0, stream>>>(g2o, g3_w, g3_b, ewb);

  // 3) fused expert trunk (WMMA + TDM)
  dim3 blk(128);
  dim3 g1g(HID / TN,  B_DIM / TM);    // 16 x 16
  dim3 g2g(HID / TN,  B_DIM / TM);    // 16 x 16
  dim3 g3g(Y_DIM / TN, B_DIM / TM);   // 10 x 16
  expert_layer_wmma<<<g1g, blk, 0, stream>>>(xh, W1h, ewb, b1, h1, nullptr,
                                             B_DIM, HID, X_MAIN, /*elu=*/1);
  expert_layer_wmma<<<g2g, blk, 0, stream>>>(h1, W2h, ewb, b2, h2, nullptr,
                                             B_DIM, HID, HID, /*elu=*/1);
  expert_layer_wmma<<<g3g, blk, 0, stream>>>(h2, W3h, ewb, b3, nullptr, y,
                                             B_DIM, Y_DIM, HID, /*elu=*/0);
}
